// MambaUpLayer_14224931685050
// MI455X (gfx1250) — matrix operand order repaired and checked by reading, not compiled, not run on hardware
//
#include <hip/hip_runtime.h>

typedef __attribute__((ext_vector_type(16))) _Float16 v16h;
typedef __attribute__((ext_vector_type(8)))  _Float16 v8h;
typedef __attribute__((ext_vector_type(8)))  float    v8f;

#define DIMC   48
#define NSTATE 16
#define DINNER 96
#define NB     2
#define SP     24
#define SPP    28                  // padded spatial (halo 1 low, 3 high for the zero tap)
#define SSRC   12
#define LSEQ   (SP*SP*SP)          // 13824
#define BLM    (NB*LSEQ)           // 27648 rows for all GEMMs
#define KCAT   144
#define KCATP  160                 // padded to 32
#define K48P   64                  // K=48 padded
#define KC2    1296                // 48*27
#define KC2P   1312                // padded to 32 (adds zero tap 27)
#define PVOX   (SPP*SPP*SPP)       // 21952

__device__ __forceinline__ float siluf(float x){ return x / (1.f + __expf(-x)); }

__device__ __forceinline__ v16h frag_a(const _Float16* p){          // halves k+0..7, k+16..23
  v8h a0 = *reinterpret_cast<const v8h*>(p);
  v8h a1 = *reinterpret_cast<const v8h*>(p + 16);
  return __builtin_shufflevector(a0, a1, 0,1,2,3,4,5,6,7,8,9,10,11,12,13,14,15);
}
__device__ __forceinline__ v16h frag_b(const _Float16* p){          // halves k+0..7, k+16..23
  v8h b0 = *reinterpret_cast<const v8h*>(p);
  v8h b1 = *reinterpret_cast<const v8h*>(p + 16);
  return __builtin_shufflevector(b0, b1, 0,1,2,3,4,5,6,7,8,9,10,11,12,13,14,15);
}

// ---------------- zero fill ----------------
__global__ void zero_f16_kernel(_Float16* __restrict__ p, int n){
  int i = blockIdx.x * blockDim.x + threadIdx.x;
  if (i < n) p[i] = (_Float16)0.f;
}

// ---------------- f32 -> f16 with (N,K) zero padding ----------------
__global__ void padcvt_f16_kernel(const float* __restrict__ s, _Float16* __restrict__ d,
                                  int Ns, int Ks, int Nd, int Kd){
  int i = blockIdx.x * blockDim.x + threadIdx.x;
  if (i >= Nd * Kd) return;
  int n = i / Kd, k = i % Kd;
  d[i] = (n < Ns && k < Ks) ? (_Float16)s[n * Ks + k] : (_Float16)0.f;
}

// ---------------- W2 reorder: dst[o][tap*48+ci] = W2[o][ci*27+tap], tap 27 = zeros ----------------
__global__ void reorder_w2_kernel(const float* __restrict__ s, _Float16* __restrict__ d){
  int i = blockIdx.x * blockDim.x + threadIdx.x;
  if (i >= DIMC * KC2P) return;
  int o = i / KC2P, k = i % KC2P;
  int tap = k / DIMC, ci = k % DIMC;
  d[i] = (tap < 27) ? (_Float16)s[(size_t)o * KC2 + ci * 27 + tap] : (_Float16)0.f;
}

// ---------------- upsample(2x, align_corners) + concat -> A1 (BLM x 160), cols 144..159 = 0 ----------------
__global__ void build_a1_kernel(const float* __restrict__ x, const float* __restrict__ Hx,
                                _Float16* __restrict__ A1){
  int idx = blockIdx.x * blockDim.x + threadIdx.x;
  if (idx >= BLM * KCATP) return;
  int c = idx % KCATP;
  int m = idx / KCATP;
  int b = m / LSEQ;
  int l = m % LSEQ;
  float v = 0.f;
  if (c < DIMC) {
    int h = l / (SP*SP), w = (l / SP) % SP, t = l % SP;
    float ph = h * 11.0f / 23.0f, pw = w * 11.0f / 23.0f, pt = t * 11.0f / 23.0f;
    int h0 = (int)ph, w0 = (int)pw, t0 = (int)pt;
    float fh = ph - h0, fw = pw - w0, ft = pt - t0;
    int h1 = min(h0 + 1, SSRC - 1), w1 = min(w0 + 1, SSRC - 1), t1 = min(t0 + 1, SSRC - 1);
    const float* xp = x + ((size_t)b * DIMC + c) * (SSRC*SSRC*SSRC);
#define XAT(a,bb,cc) xp[((a)*SSRC+(bb))*SSRC+(cc)]
    float c00 = XAT(h0,w0,t0)*(1.f-ft) + XAT(h0,w0,t1)*ft;
    float c01 = XAT(h0,w1,t0)*(1.f-ft) + XAT(h0,w1,t1)*ft;
    float c10 = XAT(h1,w0,t0)*(1.f-ft) + XAT(h1,w0,t1)*ft;
    float c11 = XAT(h1,w1,t0)*(1.f-ft) + XAT(h1,w1,t1)*ft;
#undef XAT
    v = (c00*(1.f-fw) + c01*fw)*(1.f-fh) + (c10*(1.f-fw) + c11*fw)*fh;
  } else if (c < KCAT) {
    v = Hx[((size_t)b * (2*DIMC) + (c - DIMC)) * LSEQ + l];
  }
  A1[idx] = (_Float16)v;
}

// ---------------- generic WMMA GEMM (row-major A, padded K, unconditional b128 loads) ----------------
__global__ __launch_bounds__(32)
void gemm_rm_kernel(const _Float16* __restrict__ A, int lda,
                    const _Float16* __restrict__ W, int ldw,   // Nd x ldw, ldw==Kpad
                    const float* __restrict__ bias,
                    float* __restrict__ outF, _Float16* __restrict__ outH,
                    int ldo, int N){
  const int lane  = threadIdx.x;
  const int mBase = blockIdx.x * 16;
  const int nBase = blockIdx.y * 16;
  const size_t aRow = (size_t)(mBase + (lane & 15)) * lda;
  const int nCol  = nBase + (lane & 15);
  const size_t bRow = (size_t)nCol * ldw;
  const int aKoff = (lane >> 4) * 8;
  const int bKoff = (lane >> 4) * 8;
  v8f acc = {};
  for (int kb = 0; kb < ldw; kb += 32) {
    v16h af = frag_a(A + aRow + kb + aKoff);
    v16h bf = frag_b(W + bRow + kb + bKoff);
    acc = __builtin_amdgcn_wmma_f32_16x16x32_f16(false, af, false, bf, (short)0, acc, false, false);
  }
  float bv = (bias != nullptr) ? bias[nCol < N ? nCol : 0] : 0.f;
#pragma unroll
  for (int r = 0; r < 8; ++r) {
    int m = mBase + r + ((lane >> 4) << 3);
    if (nCol < N) {
      float o = acc[r] + bv;
      if (outF) outF[(size_t)m * ldo + nCol] = o;
      if (outH) outH[(size_t)m * ldo + nCol] = (_Float16)o;
    }
  }
}

// ---------------- conv1 GEMM: epilogue scatters into zero-padded halo volume P1 ----------------
__global__ __launch_bounds__(32)
void gemm_conv1_kernel(const _Float16* __restrict__ A1, const _Float16* __restrict__ W,
                       const float* __restrict__ bias, _Float16* __restrict__ P1){
  const int lane  = threadIdx.x;
  const int mBase = blockIdx.x * 16;
  const int nBase = blockIdx.y * 16;
  const size_t aRow = (size_t)(mBase + (lane & 15)) * KCATP;
  const int nCol  = nBase + (lane & 15);
  const size_t bRow = (size_t)nCol * KCATP;
  const int aKoff = (lane >> 4) * 8;
  const int bKoff = (lane >> 4) * 8;
  v8f acc = {};
  for (int kb = 0; kb < KCATP; kb += 32) {
    v16h af = frag_a(A1 + aRow + kb + aKoff);
    v16h bf = frag_b(W + bRow + kb + bKoff);
    acc = __builtin_amdgcn_wmma_f32_16x16x32_f16(false, af, false, bf, (short)0, acc, false, false);
  }
  float bv = bias[nCol];
#pragma unroll
  for (int r = 0; r < 8; ++r) {
    int m = mBase + r + ((lane >> 4) << 3);
    int b = m / LSEQ, l = m % LSEQ;
    int h = l / (SP*SP), w = (l / SP) % SP, t = l % SP;
    size_t vox = ((size_t)b * PVOX) + ((size_t)(h+1)*SPP + (w+1))*SPP + (t+1);
    P1[vox * DIMC + nCol] = (_Float16)(acc[r] + bv);
  }
}

// ---------------- conv2 (3x3x3) GEMM: halo-volume gather, K reordered as tap*48+ci ----------------
__global__ __launch_bounds__(32)
void gemm_conv2_kernel(const _Float16* __restrict__ P1, const _Float16* __restrict__ W,
                       const float* __restrict__ bias, _Float16* __restrict__ outH){
  const int lane  = threadIdx.x;
  const int mBase = blockIdx.x * 16;
  const int nBase = blockIdx.y * 16;
  const int mRow  = mBase + (lane & 15);
  const int nCol  = nBase + (lane & 15);
  const size_t bRow = (size_t)nCol * KC2P;
  const int aKoff = (lane >> 4) * 8;
  const int bKoff = (lane >> 4) * 8;
  const int b = mRow / LSEQ;
  const int l = mRow % LSEQ;
  const int h = l / (SP*SP), w = (l / SP) % SP, t = l % SP;
  const _Float16* Pb = P1 + (size_t)b * PVOX * DIMC;
  v8f acc = {};
  for (int kb = 0; kb < KC2P; kb += 32) {
    int kk0 = kb + aKoff;
    int tap0 = kk0 / DIMC, ci0 = kk0 % DIMC;     // halo cancels the -1: index = coord + tap_digit
    int kk1 = kk0 + 16;
    int tap1 = kk1 / DIMC, ci1 = kk1 % DIMC;
    size_t v0 = ((size_t)(h + tap0/9)*SPP + (w + (tap0/3)%3))*SPP + (t + tap0%3);
    size_t v1 = ((size_t)(h + tap1/9)*SPP + (w + (tap1/3)%3))*SPP + (t + tap1%3);
    v8h a0 = *reinterpret_cast<const v8h*>(Pb + v0 * DIMC + ci0);
    v8h a1 = *reinterpret_cast<const v8h*>(Pb + v1 * DIMC + ci1);
    v16h af = __builtin_shufflevector(a0, a1, 0,1,2,3,4,5,6,7,8,9,10,11,12,13,14,15);
    v16h bf = frag_b(W + bRow + kb + bKoff);
    acc = __builtin_amdgcn_wmma_f32_16x16x32_f16(false, af, false, bf, (short)0, acc, false, false);
  }
  float bv = bias[nCol];
#pragma unroll
  for (int r = 0; r < 8; ++r) {
    int m = mBase + r + ((lane >> 4) << 3);
    outH[(size_t)m * K48P + nCol] = (_Float16)(acc[r] + bv);
  }
}

// ---------------- out_proj GEMM, epilogue: transpose + residual into d_out ----------------
__global__ __launch_bounds__(32)
void gemm_outproj_kernel(const _Float16* __restrict__ Y,   // (BLM x 96) f16
                         const _Float16* __restrict__ W,   // (48 x 96) f16
                         const float* __restrict__ Lx,     // (B,C,L) f32
                         float* __restrict__ out){
  const int lane  = threadIdx.x;
  const int mBase = blockIdx.x * 16;
  const int nBase = blockIdx.y * 16;
  const size_t aRow = (size_t)(mBase + (lane & 15)) * DINNER;
  const int nCol  = nBase + (lane & 15);
  const size_t bRow = (size_t)nCol * DINNER;
  const int aKoff = (lane >> 4) * 8;
  const int bKoff = (lane >> 4) * 8;
  v8f acc = {};
  for (int kb = 0; kb < DINNER; kb += 32) {
    v16h af = frag_a(Y + aRow + kb + aKoff);
    v16h bf = frag_b(W + bRow + kb + bKoff);
    acc = __builtin_amdgcn_wmma_f32_16x16x32_f16(false, af, false, bf, (short)0, acc, false, false);
  }
#pragma unroll
  for (int r = 0; r < 8; ++r) {
    int m = mBase + r + ((lane >> 4) << 3);
    int b = m / LSEQ, l = m % LSEQ;
    size_t oi = ((size_t)b * DIMC + nCol) * LSEQ + l;
    out[oi] = acc[r] + Lx[oi];
  }
}

// ---------------- instance-norm stats (one block per (b,c)) ----------------
__global__ __launch_bounds__(256)
void inorm_stats_kernel(const float* __restrict__ h3, float* __restrict__ stats){
  __shared__ float s1[256], s2[256];
  int bc = blockIdx.x;                 // 0..95
  int b = bc / DIMC, c = bc % DIMC;
  float sum = 0.f, ssq = 0.f;
  for (int l = threadIdx.x; l < LSEQ; l += 256) {
    float v = h3[((size_t)b * LSEQ + l) * DIMC + c];
    sum += v; ssq += v * v;
  }
  s1[threadIdx.x] = sum; s2[threadIdx.x] = ssq;
  __syncthreads();
  for (int s = 128; s > 0; s >>= 1) {
    if (threadIdx.x < s) { s1[threadIdx.x] += s1[threadIdx.x+s]; s2[threadIdx.x] += s2[threadIdx.x+s]; }
    __syncthreads();
  }
  if (threadIdx.x == 0) {
    float mean = s1[0] / (float)LSEQ;
    float var  = s2[0] / (float)LSEQ - mean * mean;
    stats[bc] = mean;
    stats[NB*DIMC + bc] = rsqrtf(var + 1e-5f);
  }
}

// ---------------- norm + SELU; Lx (B,C,L) f32, u (BLM x 64) f16 (cols 48..63 zero) ----------------
__global__ void inorm_selu_kernel(const float* __restrict__ h3, const float* __restrict__ stats,
                                  float* __restrict__ Lx, _Float16* __restrict__ u){
  int idx = blockIdx.x * blockDim.x + threadIdx.x;
  if (idx >= BLM * K48P) return;
  int c = idx % K48P;
  int m = idx / K48P;
  if (c >= DIMC) { u[idx] = (_Float16)0.f; return; }
  int b = m / LSEQ, l = m % LSEQ;
  float v = (h3[(size_t)m * DIMC + c] - stats[b*DIMC + c]) * stats[NB*DIMC + b*DIMC + c];
  const float sc = 1.0507009873554805f, al = 1.6732632423543772f;
  float r = v > 0.f ? sc * v : sc * al * (__expf(v) - 1.f);
  Lx[((size_t)b * DIMC + c) * LSEQ + l] = r;
  u[idx] = (_Float16)r;
}

// ---------------- causal depthwise conv1d + SiLU on x; SiLU(z) ----------------
__global__ void conv1d_silu_kernel(const float* __restrict__ xz,      // (BLM,192)
                                   const float* __restrict__ cw,      // (96,4)
                                   const float* __restrict__ cb,      // (96)
                                   float* __restrict__ xs32, _Float16* __restrict__ xs16,
                                   float* __restrict__ zs){
  int idx = blockIdx.x * blockDim.x + threadIdx.x;
  if (idx >= BLM * DINNER) return;
  int d = idx % DINNER;
  int m = idx / DINNER;
  int b = m / LSEQ, l = m % LSEQ;
  float acc = cb[d];
#pragma unroll
  for (int j = 0; j < 4; ++j) {
    int ll = l - 3 + j;
    if (ll >= 0) acc += cw[d*4 + j] * xz[((size_t)b * LSEQ + ll) * 192 + d];
  }
  float xs = siluf(acc);
  xs32[idx] = xs;
  xs16[idx] = (_Float16)xs;
  float z = xz[(size_t)m * 192 + DINNER + d];
  zs[idx] = siluf(z);
}

// ---------------- dt_proj + softplus ----------------
__global__ void dtproj_kernel(const float* __restrict__ dbc,  // (BLM,35)
                              const float* __restrict__ dtw,  // (96,3)
                              const float* __restrict__ dtb,  // (96)
                              float* __restrict__ delta){
  int idx = blockIdx.x * blockDim.x + threadIdx.x;
  if (idx >= BLM * DINNER) return;
  int d = idx % DINNER;
  int m = idx / DINNER;
  float a = dtb[d];
#pragma unroll
  for (int r = 0; r < 3; ++r) a += dbc[(size_t)m * 35 + r] * dtw[d*3 + r];
  delta[idx] = (a > 20.f) ? a : log1pf(__expf(a));
}

// ---------------- selective scan: thread per (b,d), 16-wide state in registers ----------------
__global__ __launch_bounds__(64)
void scan_kernel(const float* __restrict__ delta, const float* __restrict__ xs32,
                 const float* __restrict__ dbc, const float* __restrict__ zs,
                 const float* __restrict__ A_log, const float* __restrict__ Dp,
                 _Float16* __restrict__ yb){
  int idx = blockIdx.x * blockDim.x + threadIdx.x;
  if (idx >= NB * DINNER) return;
  int b = idx / DINNER, d = idx % DINNER;
  float a[NSTATE], h[NSTATE];
#pragma unroll
  for (int n = 0; n < NSTATE; ++n) { a[n] = -__expf(A_log[d*NSTATE + n]); h[n] = 0.f; }
  float Dv = Dp[d];
  for (int l = 0; l < LSEQ; ++l) {
    size_t m = (size_t)b * LSEQ + l;
    float dl = delta[m * DINNER + d];
    float xv = xs32[m * DINNER + d];
    float dx = dl * xv;
    const float* row = dbc + m * 35;
    float y = 0.f;
#pragma unroll
    for (int n = 0; n < NSTATE; ++n) {
      float hn = __expf(dl * a[n]) * h[n] + dx * row[3 + n];
      h[n] = hn;
      y += hn * row[19 + n];
    }
    y += Dv * xv;
    y *= zs[m * DINNER + d];
    yb[m * DINNER + d] = (_Float16)y;
  }
}

extern "C" void kernel_launch(void* const* d_in, const int* in_sizes, int n_in,
                              void* d_out, int out_size, void* d_ws, size_t ws_size,
                              hipStream_t stream) {
  const float* x     = (const float*)d_in[0];
  const float* Hx    = (const float*)d_in[1];
  const float* W1    = (const float*)d_in[2];
  const float* b1    = (const float*)d_in[3];
  const float* W2    = (const float*)d_in[4];
  const float* b2    = (const float*)d_in[5];
  const float* W3    = (const float*)d_in[6];
  const float* b3    = (const float*)d_in[7];
  const float* inW   = (const float*)d_in[8];
  const float* c1w   = (const float*)d_in[9];
  const float* c1b   = (const float*)d_in[10];
  const float* xW    = (const float*)d_in[11];
  const float* dtw   = (const float*)d_in[12];
  const float* dtb   = (const float*)d_in[13];
  const float* A_log = (const float*)d_in[14];
  const float* Dp    = (const float*)d_in[15];
  const float* outW  = (const float*)d_in[16];
  float* out = (float*)d_out;

  // ---- bump allocator over d_ws ----
  char* base = (char*)d_ws;
  size_t off = 0;
  auto alloc = [&](size_t bytes) -> void* {
    void* p = base + off;
    off = (off + bytes + 255) & ~(size_t)255;
    return p;
  };
  _Float16* wW1   = (_Float16*)alloc(sizeof(_Float16) * DIMC * KCATP);
  _Float16* wW2   = (_Float16*)alloc(sizeof(_Float16) * DIMC * KC2P);
  _Float16* wW3   = (_Float16*)alloc(sizeof(_Float16) * DIMC * K48P);
  _Float16* wInW  = (_Float16*)alloc(sizeof(_Float16) * 192 * K48P);
  _Float16* wXW   = (_Float16*)alloc(sizeof(_Float16) * DIMC * 96);     // rows 35..47 zero
  _Float16* wOutW = (_Float16*)alloc(sizeof(_Float16) * DIMC * 96);
  _Float16* A1    = (_Float16*)alloc(sizeof(_Float16) * (size_t)BLM * KCATP);
  _Float16* P1    = (_Float16*)alloc(sizeof(_Float16) * (size_t)NB * PVOX * DIMC);
  _Float16* act2  = (_Float16*)alloc(sizeof(_Float16) * (size_t)BLM * K48P);
  float*    h3    = (float*)   alloc(sizeof(float) * (size_t)BLM * DIMC);
  float*    stats = (float*)   alloc(sizeof(float) * 2 * NB * DIMC);
  float*    Lx    = (float*)   alloc(sizeof(float) * (size_t)BLM * DIMC);
  _Float16* u     = (_Float16*)alloc(sizeof(_Float16) * (size_t)BLM * K48P);
  float*    xz    = (float*)   alloc(sizeof(float) * (size_t)BLM * 192);
  float*    xs32  = (float*)   alloc(sizeof(float) * (size_t)BLM * DINNER);
  _Float16* xs16  = (_Float16*)alloc(sizeof(_Float16) * (size_t)BLM * DINNER);
  float*    zs    = (float*)   alloc(sizeof(float) * (size_t)BLM * DINNER);
  float*    dbc   = (float*)   alloc(sizeof(float) * (size_t)BLM * 35);
  float*    delta = (float*)   alloc(sizeof(float) * (size_t)BLM * DINNER);
  _Float16* yb    = (_Float16*)alloc(sizeof(_Float16) * (size_t)BLM * DINNER);
  (void)ws_size; (void)n_in; (void)in_sizes; (void)out_size;

  auto grid1d = [](long long n, int bs){ return (unsigned)((n + bs - 1) / bs); };

  // weight prep (padded / reordered, zero-filled)
  padcvt_f16_kernel<<<grid1d(DIMC*KCATP,256),256,0,stream>>>(W1,  wW1,  DIMC, KCAT, DIMC, KCATP);
  reorder_w2_kernel<<<grid1d(DIMC*KC2P,256),256,0,stream>>>(W2, wW2);
  padcvt_f16_kernel<<<grid1d(DIMC*K48P,256),256,0,stream>>>(W3,  wW3,  DIMC, DIMC, DIMC, K48P);
  padcvt_f16_kernel<<<grid1d(192*K48P,256),256,0,stream>>>(inW, wInW, 192,  DIMC, 192,  K48P);
  padcvt_f16_kernel<<<grid1d(DIMC*96,256),256,0,stream>>>(xW,  wXW,  35,   96,   DIMC, 96);
  padcvt_f16_kernel<<<grid1d(DIMC*96,256),256,0,stream>>>(outW,wOutW,DIMC, 96,   DIMC, 96);

  // staged inputs
  build_a1_kernel<<<grid1d((long long)BLM*KCATP,256),256,0,stream>>>(x, Hx, A1);
  zero_f16_kernel<<<grid1d((long long)NB*PVOX*DIMC,256),256,0,stream>>>(P1, NB*PVOX*DIMC);

  dim3 g48(BLM/16, 3);
  // conv1 (1x1x1): writes interior of halo volume P1
  gemm_conv1_kernel<<<g48, 32, 0, stream>>>(A1, wW1, b1, P1);
  // conv2 (3x3x3): halo gather GEMM -> act2 (ld 64, cols 48..63 zeroed by zero_f16_kernel, never written)
  zero_f16_kernel<<<grid1d((long long)BLM*K48P,256),256,0,stream>>>(act2, BLM*K48P);
  gemm_conv2_kernel<<<g48, 32, 0, stream>>>(P1, wW2, b2, act2);
  // conv3 (1x1x1): K=64 padded
  gemm_rm_kernel<<<g48, 32, 0, stream>>>(act2, K48P, wW3, K48P, b3, h3, nullptr, DIMC, DIMC);

  // instance norm + SELU
  inorm_stats_kernel<<<NB*DIMC, 256, 0, stream>>>(h3, stats);
  inorm_selu_kernel<<<grid1d((long long)BLM*K48P,256),256,0,stream>>>(h3, stats, Lx, u);

  // in_proj: (BLM x64)*(192x64)^T -> xz (BLM x192)
  dim3 g192(BLM/16, 12);
  gemm_rm_kernel<<<g192, 32, 0, stream>>>(u, K48P, wInW, K48P, nullptr, xz, nullptr, 192, 192);

  // depthwise causal conv1d + SiLU, SiLU(z)
  conv1d_silu_kernel<<<grid1d((long long)BLM*DINNER,256),256,0,stream>>>(xz, c1w, c1b, xs32, xs16, zs);

  // x_proj: (BLM x96)*(48x96)^T -> dbc (BLM x35), store-guarded to N=35
  dim3 g35(BLM/16, 3);
  gemm_rm_kernel<<<g35, 32, 0, stream>>>(xs16, DINNER, wXW, DINNER, nullptr, dbc, nullptr, 35, 35);

  // dt_proj + softplus
  dtproj_kernel<<<grid1d((long long)BLM*DINNER,256),256,0,stream>>>(dbc, dtw, dtb, delta);

  // selective scan (sequential over L, parallel over (b,d))
  scan_kernel<<<3, 64, 0, stream>>>(delta, xs32, dbc, zs, A_log, Dp, yb);

  // out_proj + transpose + residual into d_out
  gemm_outproj_kernel<<<g48, 32, 0, stream>>>(yb, wOutW, Lx, out);
}
